// CTCLoss_neg_3573412790927
// MI455X (gfx1250) — compile-verified, matrix operand order checked by reading, not run on hardware
//
#include <hip/hip_runtime.h>
#include <hip/hip_bf16.h>
#include <math.h>

#define NCROPS 10
#define LOSS_EPS 1e-4f

typedef __attribute__((ext_vector_type(16))) __bf16 v16bf;
typedef __attribute__((ext_vector_type(8)))  float  v8f;

union PackBF {
    v16bf v;
    unsigned int u[8];
};

// Round-to-nearest(-even-ish) pack of two f32 into one dword of two bf16.
__device__ __forceinline__ unsigned int pack_bf16(float lo, float hi) {
    unsigned int ulo = __float_as_uint(lo);
    unsigned int uhi = __float_as_uint(hi);
    ulo += 0x7FFFu + ((ulo >> 16) & 1u);
    uhi += 0x7FFFu + ((uhi >> 16) & 1u);
    return (ulo >> 16) | (uhi & 0xFFFF0000u);
}

// One block per batch element. 128 threads = 4 wave32s.
// Wave w accumulates K-chunks {w*32 + 128*j} of the 2x10 (teacher x student)
// dot-product tile via v_wmma_f32_16x16x32_bf16, then waves reduce in LDS and
// lane 0 runs the scalar loss epilogue.
__global__ void __launch_bounds__(128)
dino_loss_kernel(const float* __restrict__ student,
                 const float* __restrict__ teacher,
                 const int*   __restrict__ flags,
                 float* __restrict__ per_b,
                 int bsz, int d) {
    const int b    = blockIdx.x;
    const int lane = threadIdx.x & 31;
    const int wave = threadIdx.x >> 5;
    const int row  = lane & 15;   // A-matrix row (teacher crop) / B-matrix col (student crop)
    const int half = lane >> 4;   // K-halves per the 16-bit WMMA VGPR layout

    // t[b,k,:] = teacher_output[k*bsz + b, :], s[b,n,:] = student_output[n*bsz + b, :]
    const float* tRow = teacher + ((size_t)((row < 2      ? row : 0) * bsz + b)) * (size_t)d;
    const float* sRow = student + ((size_t)((row < NCROPS ? row : 0) * bsz + b)) * (size_t)d;

    // K offsets inside a 32-wide chunk for each of the 8 packed-bf16 VGPRs
    // (ISA 7.12.2, 16-bit A-matrix 16x32 layout; B mirrors it for columns).
    int kofs[8];
#pragma unroll
    for (int i = 0; i < 4; ++i) {
        kofs[i]     = half * 8 + 2 * i;
        kofs[i + 4] = 16 + half * 8 + 2 * i;
    }

    v8f acc = {};
    for (int k0 = wave * 32; k0 < d; k0 += 128) {
        PackBF A, B;
#pragma unroll
        for (int i = 0; i < 8; ++i) {
            const float2 ta = *(const float2*)(tRow + k0 + kofs[i]);
            const float2 sb = *(const float2*)(sRow + k0 + kofs[i]);
            A.u[i] = pack_bf16(ta.x, ta.y);
            B.u[i] = pack_bf16(sb.x, sb.y);
        }
        // D = A(16x32 bf16) x B(32x16 bf16) + C ; rows 0-1 / cols 0-9 are the
        // only meaningful entries, others are ignored.
        acc = __builtin_amdgcn_wmma_f32_16x16x32_bf16(
            /*neg_a=*/false, A.v, /*neg_b=*/false, B.v,
            /*c_mod=*/(short)0, acc, /*reuse_a=*/false, /*reuse_b=*/false);
    }

    // C/D layout: VGPR j, lanes 0-15 hold M=j, N=lane. We need M=0,1 (the two
    // teacher crops) at N=0..9 (student crops).
    __shared__ float red[4][2][16];
    if (lane < 16) {
        red[wave][0][lane] = acc[0];
        red[wave][1][lane] = acc[1];
    }
    __syncthreads();

    __shared__ float sLogit[2][16];
    __shared__ float sPos[16];
    if (threadIdx.x < 16) {
        const int n = threadIdx.x;
        const float dot0 = red[0][0][n] + red[1][0][n] + red[2][0][n] + red[3][0][n];
        const float dot1 = red[0][1][n] + red[1][1][n] + red[2][1][n] + red[3][1][n];
        sLogit[0][n] = expf(dot0);   // TEMP == 1.0
        sLogit[1][n] = expf(dot1);
        sPos[n] = (n < NCROPS) ? ((flags[b * NCROPS + n] != 0) ? 1.0f : 0.0f) : 0.0f;
    }
    __syncthreads();

    if (threadIdx.x == 0) {
        float npos = 0.0f;
#pragma unroll
        for (int n = 0; n < NCROPS; ++n) npos += sPos[n];

        float total = 0.0f;
#pragma unroll
        for (int k = 0; k < 2; ++k) {
            float ns = 0.0f;
#pragma unroll
            for (int n = 0; n < NCROPS; ++n)
                ns += (1.0f - sPos[n]) * sLogit[k][n];
            float lp = 0.0f;
#pragma unroll
            for (int n = 0; n < NCROPS; ++n) {
                const float lg = sLogit[k][n];
                // -log(lg / (lg + ns + eps)) = log(lg + ns + eps) - log(lg)
                lp += sPos[n] * (logf(lg + ns + LOSS_EPS) - logf(lg));
            }
            total += lp + logf(1.0f + ns + LOSS_EPS);
        }
        per_b[b] = total * 0.5f / (npos + LOSS_EPS);
    }
}

// Deterministic single-block mean over the per-batch losses.
__global__ void __launch_bounds__(256)
dino_loss_reduce(const float* __restrict__ per_b, float* __restrict__ out, int bsz) {
    __shared__ float s[256];
    float v = 0.0f;
    for (int i = threadIdx.x; i < bsz; i += 256) v += per_b[i];
    s[threadIdx.x] = v;
    __syncthreads();
    for (int st = 128; st > 0; st >>= 1) {
        if ((int)threadIdx.x < st) s[threadIdx.x] += s[threadIdx.x + st];
        __syncthreads();
    }
    if (threadIdx.x == 0) out[0] = s[0] / (float)bsz;
}

extern "C" void kernel_launch(void* const* d_in, const int* in_sizes, int n_in,
                              void* d_out, int out_size, void* d_ws, size_t ws_size,
                              hipStream_t stream) {
    const float* student = (const float*)d_in[0];   // [NCROPS*b, d] fp32
    const float* teacher = (const float*)d_in[1];   // [2*b, d]      fp32
    const int*   flags   = (const int*)d_in[2];     // [b, NCROPS]   int32

    const int bsz = in_sizes[2] / NCROPS;           // 1024
    const int d   = in_sizes[1] / (2 * bsz);        // 8192 (multiple of 128 assumed)

    float* ws = (float*)d_ws;                       // bsz floats of scratch

    dino_loss_kernel<<<bsz, 128, 0, stream>>>(student, teacher, flags, ws, bsz, d);
    dino_loss_reduce<<<1, 256, 0, stream>>>(ws, (float*)d_out, bsz);
}
